// LSTMNetworks_7834020348528
// MI455X (gfx1250) — compile-verified, matrix operand order checked by reading, not run on hardware
//
#include <hip/hip_runtime.h>

// ---------------------------------------------------------------------------
// LSTM network (peephole variant) for MI455X / gfx1250, all GEMMs via
// v_wmma_f32_16x16x32_bf16 (f32 accumulate). Recurrence exploited as
// batch-row-parallel: one workgroup owns 16 batch rows and scans all 32
// timesteps with h/c state held in LDS. Dense GEMMs use 2x2 register tiles
// per wave (4 WMMAs per fragment group) with a mov-free ping-pong pipeline
// (k-loop unrolled by 2, K must be a multiple of 64 -- holds for K=256,16384).
// ---------------------------------------------------------------------------

typedef __bf16 bf16_t;
typedef __attribute__((ext_vector_type(16))) __bf16 v16bf;
typedef __attribute__((ext_vector_type(8)))  float  v8f;

#define B_SZ 256
#define T_SZ 32
#define D_SZ 256
#define H_SZ 512
#define O_SZ 64
#define G_SZ (4 * H_SZ)       // 2048 gate width
#define BT_SZ (B_SZ * T_SZ)   // 8192
#define HT_SZ (H_SZ * T_SZ)   // 16384
#define OT_SZ (O_SZ * T_SZ)   // 2048

__device__ __forceinline__ bf16_t f2bf(float x) {
  union { float f; unsigned u; } v; v.f = x;
  unsigned r = v.u + 0x7FFFu + ((v.u >> 16) & 1u);   // round-to-nearest-even
  union { unsigned short s; bf16_t b; } o; o.s = (unsigned short)(r >> 16);
  return o.b;
}

__device__ __forceinline__ float sigmoidf_(float x) {
  return 1.0f / (1.0f + __expf(-x));
}

// Load a 16x32 bf16 WMMA fragment (A: M x K rows, or B: N x K weight rows for
// C = A @ W^T). Per ISA 16-bit fragment layout: lanes 0-15 hold K={0..7,16..23}
// of row (lane&15); lanes 16-31 hold K={8..15,24..31}. Two b128 loads per lane.
__device__ __forceinline__ v16bf load_frag(const bf16_t* __restrict__ base,
                                           int ld, int k, int lane) {
  const bf16_t* p = base + (lane & 15) * ld + k + ((lane & 16) ? 8 : 0);
  union { v16bf v; uint4 q[2]; } u;
  u.q[0] = *reinterpret_cast<const uint4*>(p);
  u.q[1] = *reinterpret_cast<const uint4*>(p + 16);
  return u.v;
}

__device__ __forceinline__ v8f wmma_bf16(v16bf a, v16bf b, v8f c) {
  return __builtin_amdgcn_wmma_f32_16x16x32_bf16(
      /*neg_a=*/false, a, /*neg_b=*/false, b,
      /*c_mod=*/(short)0, c, /*reuse_a=*/false, /*reuse_b=*/false);
}

// ---------------------------------------------------------------------------
// f32 -> bf16 conversion (vectorized x4)
// ---------------------------------------------------------------------------
__global__ __launch_bounds__(256) void k_conv(const float* __restrict__ src,
                                              bf16_t* __restrict__ dst, int n) {
  int i = (blockIdx.x * 256 + threadIdx.x) * 4;
  if (i + 3 < n) {
    float4 f = *reinterpret_cast<const float4*>(src + i);
    dst[i + 0] = f2bf(f.x); dst[i + 1] = f2bf(f.y);
    dst[i + 2] = f2bf(f.z); dst[i + 3] = f2bf(f.w);
  } else {
    for (; i < n; ++i) dst[i] = f2bf(src[i]);
  }
}

// ---------------------------------------------------------------------------
// Dense GEMM: Out[M,N] = act(A[M,K] @ W[N,K]^T + bias), f32 output.
// Block = 8 waves, block tile 32M x 256N; each wave owns a 32x32 output tile
// as 2x2 WMMA tiles. k-loop unrolled by 2 with ping-pong fragment sets so the
// software pipeline needs no inter-buffer register moves. Requires K % 64 == 0.
// ---------------------------------------------------------------------------
__global__ __launch_bounds__(256) void k_gemm(const bf16_t* __restrict__ A,
                                              const bf16_t* __restrict__ W,
                                              const float* __restrict__ bias,
                                              float* __restrict__ Out,
                                              int N, int K, int doTanh) {
  const int wave = threadIdx.x >> 5;
  const int lane = threadIdx.x & 31;
  const int m0   = blockIdx.x * 32;
  const int col  = (blockIdx.y * 8 + wave) * 32;

  const bf16_t* A0 = A + (size_t)m0 * K;
  const bf16_t* A1 = A + (size_t)(m0 + 16) * K;
  const bf16_t* W0 = W + (size_t)col * K;
  const bf16_t* W1 = W + (size_t)(col + 16) * K;

  v8f acc00 = {}, acc01 = {}, acc10 = {}, acc11 = {};

  // Ping buffers (even k-slices).
  v16bf a0 = load_frag(A0, K, 0, lane);
  v16bf a1 = load_frag(A1, K, 0, lane);
  v16bf b0 = load_frag(W0, K, 0, lane);
  v16bf b1 = load_frag(W1, K, 0, lane);

  for (int k = 0; k < K; k += 64) {
    // Pong buffers: load slice k+32 (always in range since K % 64 == 0),
    // overlapping with the WMMAs on the ping slice.
    v16bf c0 = load_frag(A0, K, k + 32, lane);
    v16bf c1 = load_frag(A1, K, k + 32, lane);
    v16bf d0 = load_frag(W0, K, k + 32, lane);
    v16bf d1 = load_frag(W1, K, k + 32, lane);
    if (k + 160 < K) {  // near-scope prefetch of the weight stream
      __builtin_prefetch(W0 + (lane & 15) * K + k + 160, 0, 3);
      __builtin_prefetch(W1 + (lane & 15) * K + k + 160, 0, 3);
    }

    acc00 = wmma_bf16(a0, b0, acc00);
    acc01 = wmma_bf16(a0, b1, acc01);
    acc10 = wmma_bf16(a1, b0, acc10);
    acc11 = wmma_bf16(a1, b1, acc11);

    // Refill ping buffers with slice k+64 while the pong WMMAs run.
    if (k + 64 < K) {
      a0 = load_frag(A0, K, k + 64, lane);
      a1 = load_frag(A1, K, k + 64, lane);
      b0 = load_frag(W0, K, k + 64, lane);
      b1 = load_frag(W1, K, k + 64, lane);
    }

    acc00 = wmma_bf16(c0, d0, acc00);
    acc01 = wmma_bf16(c0, d1, acc01);
    acc10 = wmma_bf16(c1, d0, acc10);
    acc11 = wmma_bf16(c1, d1, acc11);
  }

  const int nloc  = lane & 15;
  const int rbase = (lane & 16) ? 8 : 0;
  const float bb0 = bias ? bias[col + nloc]      : 0.0f;
  const float bb1 = bias ? bias[col + 16 + nloc] : 0.0f;
  for (int r = 0; r < 8; ++r) {
    const int row0 = m0 + rbase + r;
    const int row1 = row0 + 16;
    float v00 = acc00[r] + bb0, v01 = acc01[r] + bb1;
    float v10 = acc10[r] + bb0, v11 = acc11[r] + bb1;
    if (doTanh) { v00 = tanhf(v00); v01 = tanhf(v01);
                  v10 = tanhf(v10); v11 = tanhf(v11); }
    Out[(size_t)row0 * N + col + nloc]      = v00;
    Out[(size_t)row0 * N + col + 16 + nloc] = v01;
    Out[(size_t)row1 * N + col + nloc]      = v10;
    Out[(size_t)row1 * N + col + 16 + nloc] = v11;
  }
}

// ---------------------------------------------------------------------------
// Persistent LSTM scan. Grid = B/16 workgroups, 8 waves each. Each workgroup
// owns batch rows [b0, b0+16) and iterates t = 0..31 with h (bf16), c (f32),
// c (bf16 mirror for WMMA A-operand) in LDS. Each wave owns 4 groups of 16
// hidden columns (its columns across all four gate quarters), so gate
// nonlinearities fuse directly on WMMA accumulators.
// ---------------------------------------------------------------------------
__global__ __launch_bounds__(256) void k_scan(const float* __restrict__ obs,
                                              const float* __restrict__ b_hid,
                                              const bf16_t* __restrict__ Whid,
                                              const bf16_t* __restrict__ Wig,
                                              const bf16_t* __restrict__ Wfg,
                                              const bf16_t* __restrict__ Wog,
                                              bf16_t* __restrict__ hs) {
  __shared__ __align__(16) bf16_t h_lds[16 * H_SZ];
  __shared__ __align__(16) bf16_t cb_lds[16 * H_SZ];
  __shared__ __align__(16) float  c_lds[16 * H_SZ];

  const int wave  = threadIdx.x >> 5;
  const int lane  = threadIdx.x & 31;
  const int b0    = blockIdx.x * 16;
  const int nbase = lane & 15;
  const int mhalf = (lane & 16) ? 8 : 0;

  for (int i = threadIdx.x; i < 16 * H_SZ; i += 256) {
    h_lds[i]  = f2bf(0.0f);
    cb_lds[i] = f2bf(0.0f);
    c_lds[i]  = 0.0f;
  }
  __syncthreads();

  // Preload the per-column biases for this lane's 4 column groups x 4 quarters.
  float bh[4][4];
  for (int cg = 0; cg < 4; ++cg) {
    int colN = (wave * 4 + cg) * 16 + nbase;
    for (int q = 0; q < 4; ++q) bh[cg][q] = b_hid[q * H_SZ + colN];
  }

  v8f opre[4];  // pre-activation output gate, carried phase1 -> phase2

  for (int t = 0; t < T_SZ; ++t) {
    // ---- Phase 1: g/i/f/o gate GEMMs (A = h) + i/f peepholes (A = old c) ----
    for (int cg = 0; cg < 4; ++cg) {
      const int colH = (wave * 4 + cg) * 16;
      const bf16_t* wg  = Whid + (size_t)(0 * H_SZ + colH) * H_SZ;
      const bf16_t* wi  = Whid + (size_t)(1 * H_SZ + colH) * H_SZ;
      const bf16_t* wf  = Whid + (size_t)(2 * H_SZ + colH) * H_SZ;
      const bf16_t* wo  = Whid + (size_t)(3 * H_SZ + colH) * H_SZ;
      const bf16_t* pig = Wig + (size_t)colH * H_SZ;
      const bf16_t* pfg = Wfg + (size_t)colH * H_SZ;

      v8f aG = {}, aI = {}, aF = {}, aO = {}, pI = {}, pF = {};
      for (int k = 0; k < H_SZ; k += 32) {
        v16bf a  = load_frag(h_lds,  H_SZ, k, lane);
        v16bf cb = load_frag(cb_lds, H_SZ, k, lane);
        aG = wmma_bf16(a,  load_frag(wg,  H_SZ, k, lane), aG);
        aI = wmma_bf16(a,  load_frag(wi,  H_SZ, k, lane), aI);
        aF = wmma_bf16(a,  load_frag(wf,  H_SZ, k, lane), aF);
        aO = wmma_bf16(a,  load_frag(wo,  H_SZ, k, lane), aO);
        pI = wmma_bf16(cb, load_frag(pig, H_SZ, k, lane), pI);
        pF = wmma_bf16(cb, load_frag(pfg, H_SZ, k, lane), pF);
      }

      const int colN = colH + nbase;
      for (int r = 0; r < 8; ++r) {
        const int row = mhalf + r;
        const float* ob = obs + ((size_t)(b0 + row) * T_SZ + t) * G_SZ;
        float g = tanhf(aG[r] + ob[0 * H_SZ + colN] + bh[cg][0]);
        float i = sigmoidf_(aI[r] + ob[1 * H_SZ + colN] + bh[cg][1] + pI[r]);
        float f = sigmoidf_(aF[r] + ob[2 * H_SZ + colN] + bh[cg][2] + pF[r]);
        float cn = g * i + f * c_lds[row * H_SZ + colN];   // own column: race-free
        c_lds[row * H_SZ + colN] = cn;
        opre[cg][r] = aO[r] + ob[3 * H_SZ + colN] + bh[cg][3];
      }
    }
    __syncthreads();  // all reads of old cb_lds complete

    // ---- Refresh bf16 mirror of c for the o-peephole WMMA ----
    for (int cg = 0; cg < 4; ++cg) {
      const int colN = (wave * 4 + cg) * 16 + nbase;
      for (int r = 0; r < 8; ++r) {
        const int row = mhalf + r;
        cb_lds[row * H_SZ + colN] = f2bf(c_lds[row * H_SZ + colN]);
      }
    }
    __syncthreads();  // new c visible to all waves

    // ---- Phase 2: o-peephole (A = new c), h_new, commit h ----
    for (int cg = 0; cg < 4; ++cg) {
      const int colH = (wave * 4 + cg) * 16;
      const bf16_t* pog = Wog + (size_t)colH * H_SZ;
      v8f pO = {};
      for (int k = 0; k < H_SZ; k += 32) {
        v16bf cb = load_frag(cb_lds, H_SZ, k, lane);
        pO = wmma_bf16(cb, load_frag(pog, H_SZ, k, lane), pO);
      }
      const int colN = colH + nbase;
      for (int r = 0; r < 8; ++r) {
        const int row = mhalf + r;
        float cn = c_lds[row * H_SZ + colN];
        float h  = sigmoidf_(opre[cg][r] + pO[r]) * tanhf(cn);
        bf16_t hb = f2bf(h);
        h_lds[row * H_SZ + colN] = hb;
        hs[((size_t)(b0 + row) * T_SZ + t) * H_SZ + colN] = hb;
      }
    }
    __syncthreads();  // h complete before next step's WMMAs
  }
}

// ---------------------------------------------------------------------------
// Host-side orchestration (all on `stream`, graph-capture safe).
// ---------------------------------------------------------------------------
extern "C" void kernel_launch(void* const* d_in, const int* in_sizes, int n_in,
                              void* d_out, int out_size, void* d_ws, size_t ws_size,
                              hipStream_t stream) {
  (void)in_sizes; (void)n_in; (void)out_size; (void)ws_size;

  const float* x     = (const float*)d_in[0];
  const float* W_obs = (const float*)d_in[1];
  const float* W_hid = (const float*)d_in[2];
  const float* b_hid = (const float*)d_in[3];
  const float* W_ig  = (const float*)d_in[4];
  const float* W_fg  = (const float*)d_in[5];
  const float* W_og  = (const float*)d_in[6];
  const float* W_out = (const float*)d_in[7];
  const float* b_out = (const float*)d_in[8];
  float* out = (float*)d_out;

  // Workspace carve-up (256B aligned).
  char* ws = (char*)d_ws;
  size_t off = 0;
  auto carve = [&](size_t bytes) -> void* {
    void* p = ws + off;
    off += (bytes + 255) & ~(size_t)255;
    return p;
  };
  bf16_t* xb     = (bf16_t*)carve((size_t)BT_SZ * D_SZ * sizeof(bf16_t));
  bf16_t* Wobs_b = (bf16_t*)carve((size_t)G_SZ * D_SZ * sizeof(bf16_t));
  bf16_t* Whid_b = (bf16_t*)carve((size_t)G_SZ * H_SZ * sizeof(bf16_t));
  bf16_t* Wig_b  = (bf16_t*)carve((size_t)H_SZ * H_SZ * sizeof(bf16_t));
  bf16_t* Wfg_b  = (bf16_t*)carve((size_t)H_SZ * H_SZ * sizeof(bf16_t));
  bf16_t* Wog_b  = (bf16_t*)carve((size_t)H_SZ * H_SZ * sizeof(bf16_t));
  bf16_t* Wout_b = (bf16_t*)carve((size_t)OT_SZ * HT_SZ * sizeof(bf16_t));
  float*  obs    = (float*)carve((size_t)BT_SZ * G_SZ * sizeof(float));
  bf16_t* hsbuf  = (bf16_t*)carve((size_t)B_SZ * T_SZ * H_SZ * sizeof(bf16_t));

  auto conv = [&](const float* s, bf16_t* d, int n) {
    int blocks = (n / 4 + 255) / 256;
    k_conv<<<blocks, 256, 0, stream>>>(s, d, n);
  };
  conv(x,     xb,     BT_SZ * D_SZ);
  conv(W_obs, Wobs_b, G_SZ * D_SZ);
  conv(W_hid, Whid_b, G_SZ * H_SZ);
  conv(W_ig,  Wig_b,  H_SZ * H_SZ);
  conv(W_fg,  Wfg_b,  H_SZ * H_SZ);
  conv(W_og,  Wog_b,  H_SZ * H_SZ);
  conv(W_out, Wout_b, OT_SZ * HT_SZ);

  // obs[B*T, 4H] = x[B*T, D] @ W_obs^T   (block tile 32M x 256N)
  k_gemm<<<dim3(BT_SZ / 32, G_SZ / 256), 256, 0, stream>>>(
      xb, Wobs_b, nullptr, obs, G_SZ, D_SZ, /*tanh=*/0);

  // Recurrent scan: 16 workgroups x 16 batch rows, 32 timesteps in LDS.
  k_scan<<<dim3(B_SZ / 16), 256, 0, stream>>>(
      obs, b_hid, Whid_b, Wig_b, Wfg_b, Wog_b, hsbuf);

  // out[B, T*O] = tanh(hs[B, T*H] @ W_out^T + b_out)
  k_gemm<<<dim3(B_SZ / 32, OT_SZ / 256), 256, 0, stream>>>(
      hsbuf, Wout_b, b_out, out, OT_SZ, HT_SZ, /*tanh=*/1);
}
